// BarrierNet_1322849927855
// MI455X (gfx1250) — matrix operand order repaired and checked by reading, not compiled, not run on hardware
//
#include <hip/hip_runtime.h>
#include <hip/hip_bf16.h>

typedef _Float16 half_t;
typedef __attribute__((ext_vector_type(16))) _Float16 v16h;
typedef __attribute__((ext_vector_type(8)))  _Float16 v8h;
typedef __attribute__((ext_vector_type(2)))  _Float16 v2h;
typedef __attribute__((ext_vector_type(8)))  float    v8f;

#define OBS_X 40.0f
#define OBS_Y 15.0f
#define RADIUS 6.0f

static __device__ __forceinline__ v8f wmma_f16(v16h a, v16h b, v8f c) {
    return __builtin_amdgcn_wmma_f32_16x16x32_f16(
        /*neg_a=*/false, a, /*neg_b=*/false, b,
        /*c_mod=*/(short)0, c, /*reuse_a=*/false, /*reuse_b=*/false);
}

__global__ __launch_bounds__(256) void barriernet_kernel(
    const float* __restrict__ x,
    const float* __restrict__ mean, const float* __restrict__ stdv,
    const float* __restrict__ W1,  const float* __restrict__ b1,
    const float* __restrict__ W21, const float* __restrict__ b21,
    const float* __restrict__ W22, const float* __restrict__ b22,
    const float* __restrict__ W31, const float* __restrict__ b31,
    const float* __restrict__ W32, const float* __restrict__ b32,
    float* __restrict__ out, int nrows)
{
    __shared__ float sW1[640];           // fc1_w [128][5] fp32
    __shared__ float sb1[128];
    // B-fragments for W21^T / W22^T, pre-swizzled to exact WMMA B layout:
    // [mat][ntile][kstep][lane][elem]
    __shared__ __align__(32) half_t sFrag[2][2][4][32][16];
    // per-wave 4KB tile: first used as h1 [16][128] f16, then reused as
    // fp32 staging for x21/x22: x21 at float[0..511] ([m][n]), x22 at [512..1023]
    __shared__ __align__(32) unsigned char sTile[8][4096];

    const int t    = threadIdx.x;
    const int wave = t >> 5;
    const int lane = t & 31;
    const int r    = lane & 15;                   // row within wave's 16-row tile
    const int row  = blockIdx.x * 128 + wave * 16 + r;
    const int rowc = row < nrows ? row : (nrows - 1);

    // ---- stage weights into LDS -------------------------------------------
    for (int i = t; i < 640; i += 256) sW1[i] = W1[i];
    if (t < 128) sb1[t] = b1[t];
    {
        half_t* fr = &sFrag[0][0][0][0][0];
        #pragma unroll
        for (int rep = 0; rep < 32; ++rep) {
            int f    = t + rep * 256;             // 0..8191
            int i    =  f        & 15;
            int ln   = (f >> 4)  & 31;
            int ks   = (f >> 9)  & 3;
            int tile = (f >> 11) & 1;
            int mat  = (f >> 12) & 1;
            int k = ks * 32 + ((ln >> 4) << 3) + (i & 7) + ((i & 8) << 1);      // B layout: k = 8h+i (i<8), 16+8h+(i-8) (i>=8)
            int n = tile * 16 + (ln & 15);
            const float* Wsrc = mat ? W22 : W21;  // [32 out][128 in] row-major
            fr[f] = (half_t)Wsrc[n * 128 + k];
        }
    }
    __syncthreads();

    // ---- stage 1: x0 = x*std+mean, h1 = relu(fc1) -> LDS f16 tile ---------
    float x0[5];
    #pragma unroll
    for (int j = 0; j < 5; ++j)
        x0[j] = x[rowc * 5 + j] * stdv[j] + mean[j];
    const float px = x0[0], py = x0[1], th = x0[2], vv = x0[3];

    half_t* h1p = (half_t*)sTile[wave];           // [16][128]
    const int cb = (lane >> 4) * 64;              // lanes 0-15: cols 0-63; 16-31: 64-127
    #pragma unroll 4
    for (int cc = 0; cc < 64; cc += 2) {
        float h0 = sb1[cb + cc];
        float h1v = sb1[cb + cc + 1];
        #pragma unroll
        for (int j = 0; j < 5; ++j) {
            h0  += x0[j] * sW1[(cb + cc) * 5 + j];
            h1v += x0[j] * sW1[(cb + cc + 1) * 5 + j];
        }
        v2h p;
        p[0] = (half_t)fmaxf(h0, 0.0f);
        p[1] = (half_t)fmaxf(h1v, 0.0f);
        *(v2h*)&h1p[r * 128 + cb + cc] = p;
    }
    __syncthreads();

    // ---- stage 2: x21/x22 via WMMA (M=16, N=32, K=128) --------------------
    // A-fragment: lane holds m=r; element i: k = (i>=8?16:0) + (lane>>4)*8 + (i&7)
    v16h af[4];
    #pragma unroll
    for (int s = 0; s < 4; ++s) {
        const half_t* base = &h1p[r * 128 + s * 32 + ((lane >> 4) << 3)];
        v8h lo = *(const v8h*)(base);
        v8h hi = *(const v8h*)(base + 16);
        #pragma unroll
        for (int i = 0; i < 8; ++i) { af[s][i] = lo[i]; af[s][i + 8] = hi[i]; }
    }

    v8f a21_0 = {}, a21_1 = {}, a22_0 = {}, a22_1 = {};
    #pragma unroll
    for (int s = 0; s < 4; ++s) {
        v16h b;
        b = *(const v16h*)&sFrag[0][0][s][lane][0];  a21_0 = wmma_f16(af[s], b, a21_0);
        b = *(const v16h*)&sFrag[0][1][s][lane][0];  a21_1 = wmma_f16(af[s], b, a21_1);
        b = *(const v16h*)&sFrag[1][0][s][lane][0];  a22_0 = wmma_f16(af[s], b, a22_0);
        b = *(const v16h*)&sFrag[1][1][s][lane][0];  a22_1 = wmma_f16(af[s], b, a22_1);
    }

    // bias + relu, stage fp32 results (reuse the per-wave tile)
    const int n0 = lane & 15;
    const float bb21_0 = b21[n0], bb21_1 = b21[16 + n0];
    const float bb22_0 = b22[n0], bb22_1 = b22[16 + n0];
    float* st = (float*)sTile[wave];              // x21:[0..511], x22:[512..1023]
    const int mb = (lane >> 4) * 8;               // C/D layout: m = j + 8*(lane>=16)
    #pragma unroll
    for (int j = 0; j < 8; ++j) {
        int m = mb + j;
        st[m * 32 + n0]            = fmaxf(a21_0[j] + bb21_0, 0.0f);
        st[m * 32 + 16 + n0]       = fmaxf(a21_1[j] + bb21_1, 0.0f);
        st[512 + m * 32 + n0]      = fmaxf(a22_0[j] + bb22_0, 0.0f);
        st[512 + m * 32 + 16 + n0] = fmaxf(a22_1[j] + bb22_1, 0.0f);
    }
    __syncthreads();

    // ---- stage 3: fc31/fc32 + CBF-QP epilogue (lanes 0-15, one row each) --
    if (lane < 16 && row < nrows) {
        float a0 = b31[0], a1 = b31[1], z0 = b32[0], z1 = b32[1];
        const float* p21 = st + lane * 32;
        const float* p22 = st + 512 + lane * 32;
        #pragma unroll 8
        for (int n = 0; n < 32; ++n) {
            float q21 = p21[n], q22 = p22[n];
            a0 += q21 * W31[n];
            a1 += q21 * W31[32 + n];
            z0 += q22 * W32[n];
            z1 += q22 * W32[32 + n];
        }
        float x32_0 = 4.0f / (1.0f + expf(-z0));
        float x32_1 = 4.0f / (1.0f + expf(-z1));

        float s  = sinf(th), c = cosf(th);
        float dx = px - OBS_X, dy = py - OBS_Y;
        float barrier = dx * dx + dy * dy - RADIUS * RADIUS;
        float bdot    = 2.0f * dx * vv * c + 2.0f * dy * vv * s;
        float Lf2b    = 2.0f * vv * vv;
        float G0 = -(-2.0f * dx * vv * s + 2.0f * dy * vv * c);
        float G1 = -( 2.0f * dx * c      + 2.0f * dy * s);
        float h  = Lf2b + (x32_0 + x32_1) * bdot + x32_0 * x32_1 * barrier;
        float u00 = -a0, u01 = -a1;
        float viol = G0 * u00 + G1 * u01 - h;
        float gg   = G0 * G0 + G1 * G1;
        float lam  = fmaxf(viol, 0.0f) / (gg + 1e-12f);
        out[row * 2 + 0] = u00 - lam * G0;
        out[row * 2 + 1] = u01 - lam * G1;
    }
}

extern "C" void kernel_launch(void* const* d_in, const int* in_sizes, int n_in,
                              void* d_out, int out_size, void* d_ws, size_t ws_size,
                              hipStream_t stream) {
    (void)n_in; (void)out_size; (void)d_ws; (void)ws_size;
    const float* x    = (const float*)d_in[0];
    // d_in[1] = sgn (int scalar, unused by the reference computation)
    const float* mean = (const float*)d_in[2];
    const float* stdv = (const float*)d_in[3];
    const float* W1   = (const float*)d_in[4];
    const float* b1   = (const float*)d_in[5];
    const float* W21  = (const float*)d_in[6];
    const float* b21  = (const float*)d_in[7];
    const float* W22  = (const float*)d_in[8];
    const float* b22  = (const float*)d_in[9];
    const float* W31  = (const float*)d_in[10];
    const float* b31  = (const float*)d_in[11];
    const float* W32  = (const float*)d_in[12];
    const float* b32  = (const float*)d_in[13];
    float* out = (float*)d_out;

    int nrows = in_sizes[0] / 5;
    int grid  = (nrows + 127) / 128;   // 128 rows per block (8 waves x 16 rows)
    barriernet_kernel<<<grid, 256, 0, stream>>>(
        x, mean, stdv, W1, b1, W21, b21, W22, b22, W31, b31, W32, b32,
        out, nrows);
}
